// MultiHeadAttention_11012296147404
// MI455X (gfx1250) — matrix operand order repaired and checked by reading, not compiled, not run on hardware
//
#include <hip/hip_runtime.h>
#include <stdint.h>

// Problem constants (fixed by reference)
#define T_SEQ  2048
#define C_DIM  1024
#define NHEAD  16
#define HSZ    64
#define BATCH  4
#define BHN    (BATCH * NHEAD)       // 64
#define M_ROWS (BATCH * T_SEQ)       // 8192

typedef __attribute__((ext_vector_type(16))) __bf16          v16bf;
typedef __attribute__((ext_vector_type(8)))  __bf16          v8bf;
typedef __attribute__((ext_vector_type(8)))  float           v8f;
typedef __attribute__((ext_vector_type(8)))  unsigned short  v8us;
typedef __attribute__((ext_vector_type(4)))  int             v4i;

union FragBF { v16bf v; v8bf h[2]; };

// ---- CDNA5 async global->LDS path (guarded; fallback = sync copy) ----------
#if defined(__has_builtin)
#if __has_builtin(__builtin_amdgcn_global_load_async_to_lds_b128)
#define HAVE_ASYNC 1
#endif
#endif

#define AS1 __attribute__((address_space(1)))
#define AS3 __attribute__((address_space(3)))

__device__ __forceinline__ void async_cp16(void* lds, const void* g) {
#ifdef HAVE_ASYNC
  // param0: global (AS1) v4i*, param1: LDS (AS3) v4i*  (flat LDS addr low 32 bits
  // is the LDS offset per ISA 10.2 aperture mapping)
  __builtin_amdgcn_global_load_async_to_lds_b128(
      (AS1 v4i*)(uintptr_t)g,
      (AS3 v4i*)(unsigned)(uintptr_t)lds, 0, 0);
#else
  *(v8bf*)lds = *(const v8bf*)g;
#endif
}

template <int N>
__device__ __forceinline__ void async_wait() {
#ifdef HAVE_ASYNC
#if __has_builtin(__builtin_amdgcn_s_wait_asynccnt)
  __builtin_amdgcn_s_wait_asynccnt(N);
#else
  asm volatile("s_wait_asynccnt %0" :: "i"(N) : "memory");
#endif
#endif
}

__device__ __forceinline__ unsigned short f2bf(float f) {
  unsigned int u = __builtin_bit_cast(unsigned int, f);
  u += 0x7FFFu + ((u >> 16) & 1u);           // round-to-nearest-even
  return (unsigned short)(u >> 16);
}

// ---------------------------------------------------------------- convert
__global__ __launch_bounds__(256) void cvt_f32_bf16(
    const float* __restrict__ in, unsigned short* __restrict__ out, int n) {
  int i = (blockIdx.x * blockDim.x + threadIdx.x) * 8;
  if (i >= n) return;
  const float4 a = ((const float4*)(in + i))[0];
  const float4 c = ((const float4*)(in + i))[1];
  v8us r;
  r[0] = f2bf(a.x); r[1] = f2bf(a.y); r[2] = f2bf(a.z); r[3] = f2bf(a.w);
  r[4] = f2bf(c.x); r[5] = f2bf(c.y); r[6] = f2bf(c.z); r[7] = f2bf(c.w);
  *(v8us*)(out + i) = r;
}

// ---------------------------------------------------------------- GEMM y = A @ W^T
// A: [M_ROWS, C_DIM] bf16. W: [C_DIM, C_DIM] bf16 (row n = output col n).
// 128x128 block tile, BK=32, double-buffered LDS, 8 waves x (32x64) per wave.
// MODE 1: bf16 scattered to [B,NH,T,HS]; MODE 2: f32 [M,N] + bias.
template <int MODE>
__global__ __launch_bounds__(256) void gemm_xwT(
    const unsigned short* __restrict__ A,
    const unsigned short* __restrict__ W,
    unsigned short* __restrict__ outb,
    float* __restrict__ outf,
    const float* __restrict__ bias) {
  __shared__ __align__(16) unsigned short As[2][128 * 40];   // 128 rows x 32 k (pad 40)
  __shared__ __align__(16) unsigned short Ws[2][128 * 40];

  const int tid  = threadIdx.x;
  const int wave = tid >> 5, lane = tid & 31;
  const int half = lane >> 4, l16 = lane & 15;
  const int m0 = blockIdx.y * 128;
  const int n0 = blockIdx.x * 128;
  const int wr = (wave & 3) * 32;      // wave rows  [wr, wr+32)
  const int wc = (wave >> 2) * 64;     // wave cols  [wc, wc+64)

  const int ldrow = tid >> 1;          // 0..127
  const int ldk   = (tid & 1) * 16;    // 0 or 16

  const unsigned short* Ag = &A[(size_t)(m0 + ldrow) * C_DIM + ldk];
  const unsigned short* Wg = &W[(size_t)(n0 + ldrow) * C_DIM + ldk];
  const int lofs = ldrow * 40 + ldk;

  v8f acc[2][4];
#pragma unroll
  for (int s = 0; s < 2; ++s)
#pragma unroll
    for (int cb = 0; cb < 4; ++cb) acc[s][cb] = (v8f){};

  constexpr int KT = C_DIM / 32;
  int p = 0;

#ifdef HAVE_ASYNC
  // prime tile 0
  async_cp16(&As[0][lofs], Ag);          async_cp16(&As[0][lofs + 8], Ag + 8);
  async_cp16(&Ws[0][lofs], Wg);          async_cp16(&Ws[0][lofs + 8], Wg + 8);
#else
  {
    *(v8bf*)&As[0][lofs]     = *(const v8bf*)(Ag);
    *(v8bf*)&As[0][lofs + 8] = *(const v8bf*)(Ag + 8);
    *(v8bf*)&Ws[0][lofs]     = *(const v8bf*)(Wg);
    *(v8bf*)&Ws[0][lofs + 8] = *(const v8bf*)(Wg + 8);
  }
#endif

  for (int kt = 0; kt < KT; ++kt) {
#ifdef HAVE_ASYNC
    if (kt + 1 < KT) {   // issue next tile, then wait for current (<=4 outstanding)
      const unsigned short* An = Ag + (kt + 1) * 32;
      const unsigned short* Wn = Wg + (kt + 1) * 32;
      async_cp16(&As[p ^ 1][lofs], An);      async_cp16(&As[p ^ 1][lofs + 8], An + 8);
      async_cp16(&Ws[p ^ 1][lofs], Wn);      async_cp16(&Ws[p ^ 1][lofs + 8], Wn + 8);
      async_wait<4>();
    } else {
      async_wait<0>();
    }
    __syncthreads();
#else
    v8bf ra0, ra1, rw0, rw1;
    if (kt + 1 < KT) {   // software pipeline through registers
      ra0 = *(const v8bf*)(Ag + (kt + 1) * 32);
      ra1 = *(const v8bf*)(Ag + (kt + 1) * 32 + 8);
      rw0 = *(const v8bf*)(Wg + (kt + 1) * 32);
      rw1 = *(const v8bf*)(Wg + (kt + 1) * 32 + 8);
    }
    __syncthreads();
#endif

    // compute from buffer p: 2 A-frags x 4 B-frags -> 8 WMMAs
    FragBF af[2];
#pragma unroll
    for (int s = 0; s < 2; ++s) {
      const unsigned short* ar = &As[p][(wr + s * 16 + l16) * 40];
      af[s].h[0] = *(const v8bf*)&ar[8 * half];        // K {0..7}/{8..15}
      af[s].h[1] = *(const v8bf*)&ar[16 + 8 * half];   // K {16..23}/{24..31}
    }
#pragma unroll
    for (int cb = 0; cb < 4; ++cb) {
      FragBF bf;
      const unsigned short* br = &Ws[p][(wc + cb * 16 + l16) * 40];
      bf.h[0] = *(const v8bf*)&br[8 * half];           // K {0..7}/{8..15} of col n
      bf.h[1] = *(const v8bf*)&br[16 + 8 * half];      // K {16..23}/{24..31}
#pragma unroll
      for (int s = 0; s < 2; ++s)
        acc[s][cb] = __builtin_amdgcn_wmma_f32_16x16x32_bf16(
            false, af[s].v, false, bf.v, (short)0, acc[s][cb], false, false);
    }

#ifdef HAVE_ASYNC
    __syncthreads();     // all reads of buffer p done before it is refilled
#else
    if (kt + 1 < KT) {
      *(v8bf*)&As[p ^ 1][lofs]     = ra0;
      *(v8bf*)&As[p ^ 1][lofs + 8] = ra1;
      *(v8bf*)&Ws[p ^ 1][lofs]     = rw0;
      *(v8bf*)&Ws[p ^ 1][lofs + 8] = rw1;
    }
    __syncthreads();
#endif
    p ^= 1;
  }

#pragma unroll
  for (int s = 0; s < 2; ++s)
#pragma unroll
    for (int cb = 0; cb < 4; ++cb)
#pragma unroll
      for (int v = 0; v < 8; ++v) {
        const int m = m0 + wr + s * 16 + v + 8 * half;
        const int n = n0 + wc + cb * 16 + l16;
        if (MODE == 1) {
          const int b = m >> 11, t = m & (T_SEQ - 1);
          const int h = n >> 6, hs = n & 63;
          outb[(((size_t)(b * NHEAD + h)) * T_SEQ + t) * HSZ + hs] = f2bf(acc[s][cb][v]);
        } else {
          outf[(size_t)m * C_DIM + n] = acc[s][cb][v] + bias[n];
        }
      }
}

// ---------------------------------------------------------------- flash attention
// Q/K/V: [BHN, T, HS] bf16. Out: [B*T, C] bf16. Block = 128 queries (8 waves x 16 rows).
__global__ __launch_bounds__(256) void flash_attn(
    const unsigned short* __restrict__ Q,
    const unsigned short* __restrict__ K,
    const unsigned short* __restrict__ V,
    unsigned short* __restrict__ Ob) {
  __shared__ __align__(16) unsigned short Ks[64 * 72];        // [j][hs]
  __shared__ __align__(16) unsigned short Vs[64 * 72];        // transposed: [hs][j]
  __shared__ __align__(16) unsigned short Ps[8 * 16 * 72];    // per-wave P scratch

  const int tid  = threadIdx.x;
  const int wave = tid >> 5, lane = tid & 31;
  const int half = lane >> 4, l16 = lane & 15;
  const int bh = blockIdx.x >> 4;           // T/128 = 16 query tiles
  const int qt = blockIdx.x & 15;
  const int b = bh >> 4, h = bh & 15;
  const int q0 = qt * 128;
  const int qw0 = q0 + wave * 16;

  const unsigned short* Qb = Q + (size_t)bh * T_SEQ * HSZ;
  const unsigned short* Kb = K + (size_t)bh * T_SEQ * HSZ;
  const unsigned short* Vb = V + (size_t)bh * T_SEQ * HSZ;

  // This wave's Q rows, once: two 16x32 A-fragments (HS=64).
  FragBF qf[2];
  {
    const unsigned short* qr = &Qb[(size_t)(qw0 + l16) * HSZ];
#pragma unroll
    for (int kk = 0; kk < 2; ++kk) {
      qf[kk].h[0] = *(const v8bf*)&qr[kk * 32 + 8 * half];
      qf[kk].h[1] = *(const v8bf*)&qr[kk * 32 + 16 + 8 * half];
    }
  }

  v8f oacc[4] = {};
  float mrow[8], lrow[8];
#pragma unroll
  for (int v = 0; v < 8; ++v) { mrow[v] = -1e30f; lrow[v] = 0.0f; }

  // staging assignments
  const int kj = tid >> 2, kc = (tid & 3) * 16;        // K copy: row kj, elems [kc,kc+16)
  const int vj = (tid & 31) * 2, vh = (tid >> 5) * 8;  // V transpose: rows vj,vj+1, hs [vh,vh+8)

  const int nkv = 2 * qt + 2;               // causal: kv tiles up to the diagonal
  for (int kvt = 0; kvt < nkv; ++kvt) {
    const int kv0 = kvt * 64;

    // --- K tile: straight copy (async DMA when available) ---
    const unsigned short* kgp = &Kb[(size_t)(kv0 + kj) * HSZ + kc];
#ifdef HAVE_ASYNC
    async_cp16(&Ks[kj * 72 + kc], kgp);
    async_cp16(&Ks[kj * 72 + kc + 8], kgp + 8);
#else
    *(v8bf*)&Ks[kj * 72 + kc]     = *(const v8bf*)(kgp);
    *(v8bf*)&Ks[kj * 72 + kc + 8] = *(const v8bf*)(kgp + 8);
#endif

    // --- V tile: transposed into Vs[hs][j], packed u32 stores ---
    v8bf va = *(const v8bf*)&Vb[(size_t)(kv0 + vj) * HSZ + vh];
    v8bf vc = *(const v8bf*)&Vb[(size_t)(kv0 + vj + 1) * HSZ + vh];
#pragma unroll
    for (int e = 0; e < 8; ++e) {
      const unsigned lo = (unsigned)__builtin_bit_cast(unsigned short, va[e]);
      const unsigned hi = (unsigned)__builtin_bit_cast(unsigned short, vc[e]);
      *(unsigned*)&Vs[(vh + e) * 72 + vj] = lo | (hi << 16);
    }
    async_wait<0>();
    __syncthreads();

    if (kv0 <= qw0 + 15) {                  // wave-uniform causal skip
      const bool need_mask = (kv0 + 63 > qw0);

      // S = (Q K^T) * HS^-0.5 ; K B-frag column j = Ks row j (contiguous over hs)
      float sreg[4][8];
#pragma unroll
      for (int jb = 0; jb < 4; ++jb) {
        v8f sa = {};
        const unsigned short* kr = &Ks[(jb * 16 + l16) * 72];
#pragma unroll
        for (int kk = 0; kk < 2; ++kk) {
          FragBF bf;
          bf.h[0] = *(const v8bf*)&kr[kk * 32 + 8 * half];
          bf.h[1] = *(const v8bf*)&kr[kk * 32 + 16 + 8 * half];
          sa = __builtin_amdgcn_wmma_f32_16x16x32_bf16(
              false, qf[kk].v, false, bf.v, (short)0, sa, false, false);
        }
        const int cg = kv0 + jb * 16 + l16;
#pragma unroll
        for (int v = 0; v < 8; ++v) {
          const int rg = qw0 + v + 8 * half;
          const float s = sa[v] * 0.125f;   // HS^-0.5 = 1/8
          sreg[jb][v] = (!need_mask || cg <= rg) ? s : -1e30f;
        }
      }

      // Online softmax (row v+8*half lives across the 16 lanes of this half-wave)
#pragma unroll
      for (int v = 0; v < 8; ++v) {
        float mx = fmaxf(fmaxf(sreg[0][v], sreg[1][v]), fmaxf(sreg[2][v], sreg[3][v]));
#pragma unroll
        for (int off = 8; off >= 1; off >>= 1)
          mx = fmaxf(mx, __shfl_xor(mx, off, 16));
        const float mnew  = fmaxf(mrow[v], mx);
        const float alpha = __expf(mrow[v] - mnew);
        mrow[v] = mnew;
        float psum = 0.0f;
#pragma unroll
        for (int jb = 0; jb < 4; ++jb) {
          const float pv = __expf(sreg[jb][v] - mnew);
          sreg[jb][v] = pv;
          psum += pv;
        }
#pragma unroll
        for (int off = 8; off >= 1; off >>= 1)
          psum += __shfl_xor(psum, off, 16);
        lrow[v] = lrow[v] * alpha + psum;
#pragma unroll
        for (int hb = 0; hb < 4; ++hb)
          oacc[hb][v] *= alpha;
      }

      // Re-fragment P through per-wave LDS scratch
      unsigned short* Pw = &Ps[wave * 16 * 72];
#pragma unroll
      for (int jb = 0; jb < 4; ++jb)
#pragma unroll
        for (int v = 0; v < 8; ++v)
          Pw[(v + 8 * half) * 72 + jb * 16 + l16] = f2bf(sreg[jb][v]);
      asm volatile("s_wait_dscnt 0" ::: "memory");   // wave-local LDS RAW fence

      // O += P @ V   (V B-frag column hs = Vs row hs, contiguous over j)
#pragma unroll
      for (int hb = 0; hb < 4; ++hb) {
#pragma unroll
        for (int kk = 0; kk < 2; ++kk) {
          FragBF pa, vbf;
          const unsigned short* pr = &Pw[l16 * 72];
          pa.h[0] = *(const v8bf*)&pr[kk * 32 + 8 * half];
          pa.h[1] = *(const v8bf*)&pr[kk * 32 + 16 + 8 * half];
          const unsigned short* vr = &Vs[(hb * 16 + l16) * 72];
          vbf.h[0] = *(const v8bf*)&vr[kk * 32 + 8 * half];
          vbf.h[1] = *(const v8bf*)&vr[kk * 32 + 16 + 8 * half];
          oacc[hb] = __builtin_amdgcn_wmma_f32_16x16x32_bf16(
              false, pa.v, false, vbf.v, (short)0, oacc[hb], false, false);
        }
      }
    }
    __syncthreads();
  }

  // Epilogue: normalize and scatter to [B*T, C] bf16 for the output projection.
#pragma unroll
  for (int hb = 0; hb < 4; ++hb)
#pragma unroll
    for (int v = 0; v < 8; ++v) {
      const int rg = qw0 + v + 8 * half;
      const float o = oacc[hb][v] / lrow[v];
      Ob[((size_t)(b * T_SEQ + rg)) * C_DIM + h * HSZ + hb * 16 + l16] = f2bf(o);
    }
}

// ---------------------------------------------------------------- launch
extern "C" void kernel_launch(void* const* d_in, const int* in_sizes, int n_in,
                              void* d_out, int out_size, void* d_ws, size_t ws_size,
                              hipStream_t stream) {
  (void)in_sizes; (void)n_in; (void)out_size; (void)ws_size;
  const float* x  = (const float*)d_in[0];
  const float* Wk = (const float*)d_in[1];
  const float* Wq = (const float*)d_in[2];
  const float* Wv = (const float*)d_in[3];
  const float* Wp = (const float*)d_in[4];
  const float* bp = (const float*)d_in[5];
  float* out = (float*)d_out;

  // Workspace layout (~88 MiB total)
  char* ws = (char*)d_ws;
  const size_t szX = (size_t)M_ROWS * C_DIM * 2;   // 16 MiB (bf16 activations)
  const size_t szW = (size_t)C_DIM * C_DIM * 2;    //  2 MiB per weight
  unsigned short* xb  = (unsigned short*)(ws);
  unsigned short* wkb = (unsigned short*)(ws + szX);
  unsigned short* wqb = (unsigned short*)(ws + szX + 1 * szW);
  unsigned short* wvb = (unsigned short*)(ws + szX + 2 * szW);
  unsigned short* wpb = (unsigned short*)(ws + szX + 3 * szW);
  unsigned short* Qb  = (unsigned short*)(ws + szX + 4 * szW);
  unsigned short* Kb  = (unsigned short*)(ws + 2 * szX + 4 * szW);
  unsigned short* Vb  = (unsigned short*)(ws + 3 * szX + 4 * szW);
  unsigned short* Ab  = (unsigned short*)(ws + 4 * szX + 4 * szW);

  const int nx = M_ROWS * C_DIM;      // 8388608
  const int nw = C_DIM * C_DIM;       // 1048576
  cvt_f32_bf16<<<nx / 2048, 256, 0, stream>>>(x,  xb,  nx);
  cvt_f32_bf16<<<nw / 2048, 256, 0, stream>>>(Wk, wkb, nw);
  cvt_f32_bf16<<<nw / 2048, 256, 0, stream>>>(Wq, wqb, nw);
  cvt_f32_bf16<<<nw / 2048, 256, 0, stream>>>(Wv, wvb, nw);
  cvt_f32_bf16<<<nw / 2048, 256, 0, stream>>>(Wp, wpb, nw);

  dim3 gg(C_DIM / 128, M_ROWS / 128);   // 8 x 64 blocks
  gemm_xwT<1><<<gg, 256, 0, stream>>>(xb, wqb, Qb, nullptr, nullptr);
  gemm_xwT<1><<<gg, 256, 0, stream>>>(xb, wkb, Kb, nullptr, nullptr);
  gemm_xwT<1><<<gg, 256, 0, stream>>>(xb, wvb, Vb, nullptr, nullptr);

  flash_attn<<<BHN * (T_SEQ / 128), 256, 0, stream>>>(Qb, Kb, Vb, Ab);

  gemm_xwT<2><<<gg, 256, 0, stream>>>(Ab, wpb, nullptr, out, bp);
}
